// VolumeSDFRenderer_11020886082110
// MI455X (gfx1250) — matrix operand order repaired and checked by reading, not compiled, not run on hardware
//
#include <hip/hip_runtime.h>

// ---------------------------------------------------------------------------
// VolSDF volume rendering on MI455X (gfx1250, wave32).
//
// Bandwidth-bound (168 MB @ 23.3 TB/s => ~7.2 us floor). Single pass, every
// byte read once with b128 loads. The length-128 exclusive scan of
// t = -sigma*delta is done as a matmul-scan on the WMMA unit:
//   P(16x16) = L(16x32,strictly-lower-tri) x Tmat(32x16) + carry
// using v_wmma_f32_16x16x32_f16, 2 WMMAs per 32-sample chunk, 4 chunks,
// 16 rays per wave32 (matches the 16x16 C/D tile). Carry chain stays f32.
// Weights via w_m = T_m - T_{m+1}; RGB dot product is lane-local in D layout.
// ---------------------------------------------------------------------------

typedef __attribute__((ext_vector_type(16))) _Float16 v16h;
typedef __attribute__((ext_vector_type(8)))  float    v8f;

#define ALPHA_C   10.0f
#define INV_BETA  20.0f      // 1/0.05
#define EPS_C     1e-10f
#define FARDELTA  1e10f
#define DD        128

__device__ __forceinline__ float laplace_density(float sd) {
  // alpha * (s<=0 ? 0.5*exp(s/b) : 1-0.5*exp(-s/b)), s = -sd
  float s = -sd;
  float e = __expf(-fabsf(s) * INV_BETA);
  return (s <= 0.0f) ? (ALPHA_C * 0.5f * e) : (ALPHA_C * (1.0f - 0.5f * e));
}

__global__ __launch_bounds__(256)
void volsdf_wmma_scan_kernel(const float* __restrict__ sd,
                             const float* __restrict__ color,
                             const float* __restrict__ depth,
                             float* __restrict__ out) {
  const int lane  = threadIdx.x & 31;
  const int gwave = blockIdx.x * (blockDim.x >> 5) + (threadIdx.x >> 5);
  const int n     = lane & 15;   // ray column within tile (== A-row index)
  const int h     = lane >> 4;   // half-wave select
  const int ray   = gwave * 16 + n;

  const float* sdr = sd    + (size_t)ray * DD;
  const float* der = depth + (size_t)ray * DD;
  const float* cor = color + (size_t)ray * DD * 3;

  // --- Constant A matrices (strictly lower-triangular ones), this code's 16-bit A
  // 16x32 layout: lane m=lane&15; elem e -> K = 16h + e (contiguous, matches B).
  v16h a1, a2;
#pragma unroll
  for (int e = 0; e < 16; ++e) {
    const int k = 16 * h + e;
    a1[e] = (_Float16)((k < n)      ? 1.0f : 0.0f);  // prefixes m = 0..15
    a2[e] = (_Float16)((k < n + 16) ? 1.0f : 0.0f);  // prefixes m = 16..31
  }

  float carry = 0.0f;                 // running exclusive cumsum of t (f32)
  float accR = 0.0f, accG = 0.0f, accB = 0.0f;

#pragma unroll
  for (int c = 0; c < 4; ++c) {
    const int cbase = 32 * c;

    // ---- Build B (32x16, lane holds column n, K = 16h + j contiguous) ----
    const int base = cbase + 16 * h;            // 16 consecutive samples
    float4 s4[4], d4[4];
#pragma unroll
    for (int q = 0; q < 4; ++q) {
      s4[q] = *(const float4*)(sdr + base + 4 * q);
      d4[q] = *(const float4*)(der + base + 4 * q);
    }
    const int   i16  = (base + 16 < DD) ? (base + 16) : (DD - 1);
    const float d16v = der[i16];
    const float dep[17] = { d4[0].x, d4[0].y, d4[0].z, d4[0].w,
                            d4[1].x, d4[1].y, d4[1].z, d4[1].w,
                            d4[2].x, d4[2].y, d4[2].z, d4[2].w,
                            d4[3].x, d4[3].y, d4[3].z, d4[3].w, d16v };
    const float sdv[16] = { s4[0].x, s4[0].y, s4[0].z, s4[0].w,
                            s4[1].x, s4[1].y, s4[1].z, s4[1].w,
                            s4[2].x, s4[2].y, s4[2].z, s4[2].w,
                            s4[3].x, s4[3].y, s4[3].z, s4[3].w };
    v16h bmat;
    float t15 = 0.0f;                           // t[31] lives on h==1, j==15
#pragma unroll
    for (int j = 0; j < 16; ++j) {
      const int   s     = base + j;
      const float delta = (s == DD - 1) ? FARDELTA : (dep[j + 1] - dep[j]);
      const float t     = -laplace_density(sdv[j]) * delta + EPS_C;
      if (j == 15) t15 = t;
      // clamp sentinel so 0 x (-inf) cannot NaN-poison the WMMA
      bmat[j] = (_Float16)fmaxf(t, -1.0e4f);
    }

    // ---- Matmul-scan: exclusive prefixes + carry broadcast via C ----
    v8f cvec;
#pragma unroll
    for (int j = 0; j < 8; ++j) cvec[j] = carry;

    v8f p1 = __builtin_amdgcn_wmma_f32_16x16x32_f16(
        false, a1, false, bmat, (short)0, cvec, false, false);
    v8f p2 = __builtin_amdgcn_wmma_f32_16x16x32_f16(
        false, a2, false, bmat, (short)0, cvec, false, false);

    // ---- Carry update: on h==1, p2[7] = carry + sum_{k<31} t_k ----
    const float tot = p2[7] + t15;                 // valid on h==1 lanes
    carry = __shfl(tot, n + 16, 32);               // broadcast to both halves

    // ---- Boundary prefix values from the other half-wave ----
    const float bp1 = __shfl(p1[0], lane ^ 16, 32);  // h0: logT[8]  h1: logT[0]
    const float bp2 = __shfl(p2[0], lane ^ 16, 32);  // h0: logT[24] h1: logT[16]
    const float next1 = (h == 0) ? bp1 : bp2;        // logT at m = 8h+8
    const float next2 = (h == 0) ? bp2 : carry;      // logT at m = 16+8h+8

    // ---- Weights w_m = T_m - T_{m+1}; color in D layout (m = 16g+8h+j) ----
#pragma unroll
    for (int g = 0; g < 2; ++g) {
      float T[9];
#pragma unroll
      for (int j = 0; j < 8; ++j) T[j] = __expf(g ? p2[j] : p1[j]);
      T[8] = __expf(g ? next2 : next1);

      const float* cp = cor + (size_t)(cbase + 16 * g + 8 * h) * 3;
      float4 cc[6];
#pragma unroll
      for (int q = 0; q < 6; ++q) cc[q] = ((const float4*)cp)[q];
      const float* cf = (const float*)cc;
#pragma unroll
      for (int j = 0; j < 8; ++j) {
        const float w = T[j] - T[j + 1];
        accR = fmaf(w, cf[3 * j + 0], accR);
        accG = fmaf(w, cf[3 * j + 1], accG);
        accB = fmaf(w, cf[3 * j + 2], accB);
      }
    }
  }

  // ---- Reduce the two half-wave partials, lane h==0 writes the ray ----
  accR += __shfl_xor(accR, 16, 32);
  accG += __shfl_xor(accG, 16, 32);
  accB += __shfl_xor(accB, 16, 32);
  if (h == 0) {
    float* o = out + (size_t)ray * 3;
    o[0] = accR; o[1] = accG; o[2] = accB;
  }
}

extern "C" void kernel_launch(void* const* d_in, const int* in_sizes, int n_in,
                              void* d_out, int out_size, void* d_ws, size_t ws_size,
                              hipStream_t stream) {
  const float* sd    = (const float*)d_in[0];  // [N,128]
  const float* color = (const float*)d_in[1];  // [N,128,3]
  const float* depth = (const float*)d_in[2];  // [N,128]
  float* out = (float*)d_out;                  // [N,3]

  const int N = in_sizes[0] / DD;              // 65536 rays
  // 16 rays per wave32, 8 waves per 256-thread block -> 128 rays/block.
  const int raysPerBlock = 128;
  dim3 grid((N + raysPerBlock - 1) / raysPerBlock), block(256);
  volsdf_wmma_scan_kernel<<<grid, block, 0, stream>>>(sd, color, depth, out);
}
